// AttentionBlock_24988119728335
// MI455X (gfx1250) — matrix operand order repaired and checked by reading, not compiled, not run on hardware
//
#include <hip/hip_runtime.h>

#define B_  2
#define S_  2048
#define D_  1024
#define H_  16
#define DK_ 64

typedef __attribute__((ext_vector_type(16))) __bf16 bf16x16;
typedef __attribute__((ext_vector_type(8)))  float  f32x8;

union FragBF { bf16x16 v; unsigned short u[16]; };

__device__ __forceinline__ unsigned short f2bf(float f) {
  unsigned int u = __float_as_uint(f);
  u += 0x7FFFu + ((u >> 16) & 1u);   // round-to-nearest-even
  return (unsigned short)(u >> 16);
}

__device__ __forceinline__ f32x8 zero8() {
  f32x8 z;
#pragma unroll
  for (int i = 0; i < 8; ++i) z[i] = 0.0f;
  return z;
}

// A fragment: 16 rows (M) x 32 (K), bf16, row-major source with leading dim ld.
__device__ __forceinline__ bf16x16 load_frag_a(const unsigned short* p, int ld, int lane) {
  const int m  = lane & 15;
  const int kh = (lane >> 4) & 1;
  const unsigned short* r = p + m * ld + kh * 16;
  FragBF f;
#pragma unroll
  for (int j = 0; j < 16; ++j) f.u[j] = r[j];
  return f.v;
}

// B fragment: 32 (K) x 16 cols (N). Source tile stored as B^T rows: tile[n][k].
__device__ __forceinline__ bf16x16 load_frag_b(const unsigned short* p, int ld, int lane) {
  const int n  = lane & 15;
  const int kh = (lane >> 4) & 1;
  const unsigned short* r = p + n * ld + kh * 16;
  FragBF f;
#pragma unroll
  for (int j = 0; j < 16; ++j) f.u[j] = r[j];
  return f.v;
}

__device__ __forceinline__ f32x8 wmma_bf16(bf16x16 a, bf16x16 b, f32x8 c) {
  return __builtin_amdgcn_wmma_f32_16x16x32_bf16(false, a, false, b, (short)0, c, false, false);
}

// gfx1250 async copy: 16B global -> LDS, tracked by ASYNCcnt (no VGPR staging).
__device__ __forceinline__ void async_copy16(unsigned lds_addr, const void* gaddr) {
  asm volatile("global_load_async_to_lds_b128 %0, %1, off"
               :: "v"(lds_addr), "v"((unsigned long long)(uintptr_t)gaddr)
               : "memory");
}
__device__ __forceinline__ void wait_asynccnt_le8() {
  asm volatile("s_wait_asynccnt 8" ::: "memory");
}
__device__ __forceinline__ void wait_asynccnt_0() {
  asm volatile("s_wait_asynccnt 0" ::: "memory");
}
__device__ __forceinline__ unsigned lds_addr32(const void* p) {
  return (unsigned)(uintptr_t)p;    // flat->LDS uses addr[31:0] (ISA 10.2)
}

// ---------------------------------------------------------------------------
// C[M=4096, N=1024] = A[4096,1024] @ W^T + bias. fp32 sources, bf16 compute.
// mode 0: out bf16 [B,H,S,DK]   (Q,K)
// mode 1: out bf16 [B,H,DK,S]   (V transposed)
// mode 2: out fp32 [M, D]       (final projection)
// ---------------------------------------------------------------------------
__global__ __launch_bounds__(128)
void proj_gemm(const float* __restrict__ A, const float* __restrict__ W,
               const float* __restrict__ bias, void* __restrict__ out, int mode)
{
  __shared__ __align__(16) unsigned short sA[64 * 40];
  __shared__ __align__(16) unsigned short sB[64 * 40];

  const int tid  = threadIdx.x;
  const int lane = tid & 31;
  const int wave = tid >> 5;            // 4 waves
  const int wr   = (wave >> 1) * 32;    // wave's 32x32 quadrant
  const int wc   = (wave & 1) * 32;
  const int m0   = blockIdx.x * 64;
  const int n0   = blockIdx.y * 64;

  f32x8 acc[2][2];
#pragma unroll
  for (int i = 0; i < 2; ++i)
#pragma unroll
    for (int j = 0; j < 2; ++j) acc[i][j] = zero8();

  const int srow = tid >> 1;            // staging: 64 rows x 32 cols, 2 thr/row
  const int scol = (tid & 1) * 16;

  for (int k0 = 0; k0 < D_; k0 += 32) {
    __syncthreads();
    {
      const float* ga = A + (size_t)(m0 + srow) * D_ + k0 + scol;
      const float* gb = W + (size_t)(n0 + srow) * D_ + k0 + scol;
      unsigned short* da = sA + srow * 40 + scol;
      unsigned short* db = sB + srow * 40 + scol;
#pragma unroll
      for (int i = 0; i < 16; i += 4) {
        float4 fa = *(const float4*)(ga + i);
        da[i] = f2bf(fa.x); da[i+1] = f2bf(fa.y); da[i+2] = f2bf(fa.z); da[i+3] = f2bf(fa.w);
        float4 fb = *(const float4*)(gb + i);
        db[i] = f2bf(fb.x); db[i+1] = f2bf(fb.y); db[i+2] = f2bf(fb.z); db[i+3] = f2bf(fb.w);
      }
    }
    __syncthreads();

    bf16x16 a0 = load_frag_a(sA + (wr     ) * 40, 40, lane);
    bf16x16 a1 = load_frag_a(sA + (wr + 16) * 40, 40, lane);
    bf16x16 b0 = load_frag_b(sB + (wc     ) * 40, 40, lane);
    bf16x16 b1 = load_frag_b(sB + (wc + 16) * 40, 40, lane);

    acc[0][0] = wmma_bf16(a0, b0, acc[0][0]);
    acc[0][1] = wmma_bf16(a0, b1, acc[0][1]);
    acc[1][0] = wmma_bf16(a1, b0, acc[1][0]);
    acc[1][1] = wmma_bf16(a1, b1, acc[1][1]);
  }

  const int half = lane >> 4;
  const int nl   = lane & 15;
#pragma unroll
  for (int i = 0; i < 2; ++i) {
#pragma unroll
    for (int j = 0; j < 2; ++j) {
#pragma unroll
      for (int r = 0; r < 8; ++r) {
        const int m = m0 + wr + i * 16 + r + 8 * half;  // C layout: row = r + 8*half
        const int n = n0 + wc + j * 16 + nl;            // col = lane&15
        const float val = acc[i][j][r] + bias[n];
        if (mode == 2) {
          ((float*)out)[(size_t)m * D_ + n] = val;
        } else {
          const int b = m >> 11, s = m & (S_ - 1);      // M = b*S + s
          const int h = n >> 6,  dk = n & (DK_ - 1);    // N = h*DK + dk
          if (mode == 0)
            ((unsigned short*)out)[((((size_t)b * H_ + h) * S_ + s) << 6) + dk] = f2bf(val);
          else
            ((unsigned short*)out)[((((size_t)b * H_ + h) * DK_ + dk) << 11) + s] = f2bf(val);
        }
      }
    }
  }
}

// ---------------------------------------------------------------------------
// Flash-style causal attention. Q,K: bf16 [B,H,S,DK]; Vt: bf16 [B,H,DK,S].
// Output: fp32 [B,S,D]. 64 query rows / block, 4 waves x 16 rows; key blocks
// of 64 with online softmax. K/V tiles double-buffered in LDS via
// GLOBAL_LOAD_ASYNC_TO_LDS_B128 (ASYNCcnt), prefetched one block ahead.
// ---------------------------------------------------------------------------
__global__ __launch_bounds__(128)
void attn_fwd(const unsigned short* __restrict__ Q, const unsigned short* __restrict__ K,
              const unsigned short* __restrict__ Vt, float* __restrict__ O)
{
  __shared__ __align__(16) unsigned short sK[2][64 * 72];    // [buf][key][dk]
  __shared__ __align__(16) unsigned short sV[2][64 * 72];    // [buf][dk][key]
  __shared__ __align__(16) unsigned short sP[4 * 16 * 72];   // per-wave P scratch

  const int tid  = threadIdx.x;
  const int lane = tid & 31;
  const int wave = tid >> 5;
  const int bh   = blockIdx.y;          // b*H + h
  const int qb   = blockIdx.x * 64;
  const int qrow = qb + wave * 16;

  const unsigned short* Qh = Q  + (size_t)bh * S_ * DK_;
  const unsigned short* Kh = K  + (size_t)bh * S_ * DK_;
  const unsigned short* Vh = Vt + (size_t)bh * DK_ * S_;

  // Q fragments for this wave's 16 rows (K-dim 64 -> two 32-steps)
  bf16x16 qa0 = load_frag_a(Qh + (size_t)qrow * DK_ +  0, DK_, lane);
  bf16x16 qa1 = load_frag_a(Qh + (size_t)qrow * DK_ + 32, DK_, lane);

  unsigned short* sPw = sP + wave * 16 * 72;

  f32x8 o[4];
  float mstate[8], lstate[8];
#pragma unroll
  for (int j = 0; j < 4; ++j) o[j] = zero8();
#pragma unroll
  for (int r = 0; r < 8; ++r) { mstate[r] = -1e30f; lstate[r] = 0.0f; }

  const int half = lane >> 4;
  const int nl   = lane & 15;
  const int srow = tid >> 1;            // staging: 64 rows, 2 threads/row
  const int sc0  = (tid & 1) * 32;      // each thread: 32 halves = 4x16B

  const int nkb = blockIdx.x + 1;       // causal: keys <= qb+63

  // issue async prefetch of key-block kb into LDS buffer `buf` (8 ops/thread)
  auto prefetch = [&](int kb, int buf) {
    const char* gk = (const char*)(Kh + (size_t)(kb + srow) * DK_ + sc0);
    const char* gv = (const char*)(Vh + (size_t)srow * S_ + kb + sc0);
    unsigned lk = lds_addr32(&sK[buf][srow * 72 + sc0]);
    unsigned lv = lds_addr32(&sV[buf][srow * 72 + sc0]);
#pragma unroll
    for (int i = 0; i < 4; ++i) {
      async_copy16(lk + i * 16, gk + i * 16);
      async_copy16(lv + i * 16, gv + i * 16);
    }
  };

  prefetch(0, 0);

  for (int kbi = 0; kbi < nkb; ++kbi) {
    const int kb  = kbi * 64;
    const int cur = kbi & 1;

    __syncthreads();                    // buffer cur^1 free (last read in previous iter)
    if (kbi + 1 < nkb) {
      prefetch(kb + 64, cur ^ 1);       // overlap next tile with this block's math
      wait_asynccnt_le8();              // oldest 8 (= tile `cur`) have landed
    } else {
      wait_asynccnt_0();
    }
    __syncthreads();                    // tile `cur` visible to all waves

    const unsigned short* sKc = sK[cur];
    const unsigned short* sVc = sV[cur];

    // scores S = (Q K^T) * 1/sqrt(DK), causal mask
    float sc[4][8];
#pragma unroll
    for (int kn = 0; kn < 4; ++kn) {
      f32x8 s = zero8();
      bf16x16 b0 = load_frag_b(sKc + (kn * 16) * 72 +  0, 72, lane);
      bf16x16 b1 = load_frag_b(sKc + (kn * 16) * 72 + 32, 72, lane);
      s = wmma_bf16(qa0, b0, s);
      s = wmma_bf16(qa1, b1, s);
      const int key = kb + kn * 16 + nl;
#pragma unroll
      for (int r = 0; r < 8; ++r) {
        const int qg = qrow + r + 8 * half;
        sc[kn][r] = (key > qg) ? -1e30f : s[r] * 0.125f;
      }
    }

    // online softmax per row (row r+8*half replicated across 16 lanes)
#pragma unroll
    for (int r = 0; r < 8; ++r) {
      float mx = sc[0][r];
#pragma unroll
      for (int kn = 1; kn < 4; ++kn) mx = fmaxf(mx, sc[kn][r]);
#pragma unroll
      for (int off = 1; off < 16; off <<= 1) mx = fmaxf(mx, __shfl_xor(mx, off, 32));
      const float mo = mstate[r];
      const float mn = fmaxf(mo, mx);
      const float alpha = __expf(mo - mn);
      float ls = 0.0f;
#pragma unroll
      for (int kn = 0; kn < 4; ++kn) {
        const float p = __expf(sc[kn][r] - mn);
        ls += p;
        sPw[(r + 8 * half) * 72 + kn * 16 + nl] = f2bf(p);   // C-layout -> LDS
      }
#pragma unroll
      for (int off = 1; off < 16; off <<= 1) ls += __shfl_xor(ls, off, 32);
      lstate[r] = lstate[r] * alpha + ls;
      mstate[r] = mn;
#pragma unroll
      for (int j = 0; j < 4; ++j) o[j][r] *= alpha;          // rescale accum row
    }

    // wave-local LDS round trip: P stores must land before A-frag reads
    asm volatile("s_wait_dscnt 0" ::: "memory");

    bf16x16 pa0 = load_frag_a(sPw +  0, 72, lane);
    bf16x16 pa1 = load_frag_a(sPw + 32, 72, lane);
#pragma unroll
    for (int jn = 0; jn < 4; ++jn) {
      bf16x16 bv0 = load_frag_b(sVc + (jn * 16) * 72 +  0, 72, lane);
      bf16x16 bv1 = load_frag_b(sVc + (jn * 16) * 72 + 32, 72, lane);
      o[jn] = wmma_bf16(pa0, bv0, o[jn]);
      o[jn] = wmma_bf16(pa1, bv1, o[jn]);
    }
  }

  // normalize and write heads back into [B, S, D] fp32
  const int b = bh >> 4;
  const int h = bh & 15;
#pragma unroll
  for (int jn = 0; jn < 4; ++jn) {
#pragma unroll
    for (int r = 0; r < 8; ++r) {
      const int s = qrow + r + 8 * half;
      const int c = h * DK_ + jn * 16 + nl;
      O[((size_t)b * S_ + s) * D_ + c] = o[jn][r] / lstate[r];
    }
  }
}

// ---------------------------------------------------------------------------
extern "C" void kernel_launch(void* const* d_in, const int* in_sizes, int n_in,
                              void* d_out, int out_size, void* d_ws, size_t ws_size,
                              hipStream_t stream) {
  (void)in_sizes; (void)n_in; (void)out_size; (void)ws_size;
  const float* query  = (const float*)d_in[0];
  const float* key_in = (const float*)d_in[1];
  const float* value  = (const float*)d_in[2];
  // d_in[3] = mask (causal, handled analytically)
  const float* Wq_w = (const float*)d_in[4];
  const float* Wq_b = (const float*)d_in[5];
  const float* Wk_w = (const float*)d_in[6];
  const float* Wk_b = (const float*)d_in[7];
  const float* Wv_w = (const float*)d_in[8];
  const float* Wv_b = (const float*)d_in[9];
  const float* Wo_w = (const float*)d_in[10];
  const float* Wo_b = (const float*)d_in[11];

  unsigned char* ws = (unsigned char*)d_ws;
  unsigned short* Qws  = (unsigned short*)(ws);                              // 8 MB bf16 [B,H,S,DK]
  unsigned short* Kws  = (unsigned short*)(ws + (size_t)8  * 1024 * 1024);   // 8 MB bf16 [B,H,S,DK]
  unsigned short* Vtws = (unsigned short*)(ws + (size_t)16 * 1024 * 1024);   // 8 MB bf16 [B,H,DK,S]
  float*          Xws  = (float*)        (ws + (size_t)24 * 1024 * 1024);    // 16 MB fp32 [B,S,D]

  const dim3 gproj(64, 16), blk(128);
  proj_gemm<<<gproj, blk, 0, stream>>>(query,  Wq_w, Wq_b, Qws,  0);
  proj_gemm<<<gproj, blk, 0, stream>>>(key_in, Wk_w, Wk_b, Kws,  0);
  proj_gemm<<<gproj, blk, 0, stream>>>(value,  Wv_w, Wv_b, Vtws, 1);
  attn_fwd<<<dim3(32, 32), blk, 0, stream>>>(Qws, Kws, Vtws, Xws);
  proj_gemm<<<gproj, blk, 0, stream>>>(Xws, Wo_w, Wo_b, d_out, 2);
}
